// MoEFusion_40432822124690
// MI455X (gfx1250) — matrix operand order repaired and checked by reading, not compiled, not run on hardware
//
#include <hip/hip_runtime.h>
#include <hip/hip_bf16.h>
#include <math.h>

// ---------------------------------------------------------------------------
// MoE fusion for MI455X (gfx1250, wave32, WMMA f32_16x16x32_f16)
// N=8192 tokens, E=768, 2E=1536, 4E=3072, NE=12 experts, TOPK=2
// gate -> top2 routing -> per-expert sparse MLP (6x FLOP cut) ->
// weighted scatter + combine.  Heavy GEMMs via V_WMMA_F32_16X16X32_F16,
// BK=64 double K-step (16 WMMAs per barrier phase), packed f16 LDS writes.
// ---------------------------------------------------------------------------

#define TOK_N   8192
#define DIM_E   768
#define DIM_2E  1536
#define DIM_4E  3072
#define NEXP    12

typedef _Float16 h2pack __attribute__((ext_vector_type(2)));
typedef _Float16 half8  __attribute__((ext_vector_type(8)));
typedef _Float16 half16 __attribute__((ext_vector_type(16)));
typedef float    v8f    __attribute__((ext_vector_type(8)));

union AFrag { half16 v; half8 h[2]; };
union F4 { float4 v; float s[4]; };

__device__ __forceinline__ float gelu_exact(float x) {
    return 0.5f * x * (1.0f + erff(x * 0.70710678118654752440f));
}

__device__ __forceinline__ half16 cvt16(const float4* p) {
    half16 hv;
    #pragma unroll
    for (int j = 0; j < 4; ++j) {
        float4 f = p[j];
        hv[j * 4 + 0] = (_Float16)f.x;
        hv[j * 4 + 1] = (_Float16)f.y;
        hv[j * 4 + 2] = (_Float16)f.z;
        hv[j * 4 + 3] = (_Float16)f.w;
    }
    return hv;
}

// ---------------------------------------------------------------------------
// 128x128 output tile, BK=64, 256 threads (8 waves: 2 M-waves x 4 N-waves).
// MODE 0: gate layer1   : A = combined(f32)[row=global],  out = gelu -> f16 gh
// MODE 1: expert layer1 : A = combined(f32)[row=gather],  out = gelu -> f16 H
// MODE 2: expert layer2 : A = H(f16)[row=local],          out = sigmoid*w -> f32 slot
// ---------------------------------------------------------------------------
template <int MODE>
__global__ __launch_bounds__(256)
void moe_gemm_kernel(const float* __restrict__ Af32,
                     const _Float16* __restrict__ Af16,
                     const float* __restrict__ W,      // [K, Ncols] row-major
                     const float* __restrict__ bias,   // [Ncols]
                     _Float16* __restrict__ Out16,
                     float* __restrict__ Out32,
                     const int* __restrict__ entries,  // [NEXP*TOK_N] token*2+slot
                     const float* __restrict__ entry_w,
                     const int* __restrict__ counts,
                     int expert, int M_fixed, int K, int Ncols)
{
    const int tid  = threadIdx.x;
    const int lane = tid & 31;
    const int wid  = tid >> 5;
    const int wm   = wid >> 2;        // 0..1  (64 rows per M-wave)
    const int wn   = wid & 3;         // 0..3  (32 cols per N-wave)
    const int n0   = blockIdx.x * 128;
    const int m0   = blockIdx.y * 128;

    int cnt = M_fixed;
    if (MODE != 0) cnt = counts[expert];
    if (m0 >= cnt) return;            // uniform early exit for empty tiles

    __shared__ __align__(16) _Float16 As[128 * 64]; // row-major [m][k]   16KB
    __shared__ __align__(16) _Float16 Bs[128 * 64]; // col-major [n][k]   16KB

    v8f acc[4][2] = {};
    const int ebase = expert * TOK_N;

    // ---- per-thread cooperative-load coordinates (loop-invariant) ---------
    const int arow_ld = tid >> 1;             // A: 0..127
    const int akseg   = (tid & 1) * 32;       // A: 0 or 32
    const int alr     = m0 + arow_ld;
    const bool aval   = alr < cnt;
    size_t asrc = 0;
    if (MODE == 1 && aval) asrc = (size_t)(entries[ebase + alr] >> 1);
    else                   asrc = (size_t)alr;

    const int bkq   = tid >> 4;               // B: k-quad id 0..15
    const int bcol8 = (tid & 15) * 8;         // B: 8 columns

    for (int k0 = 0; k0 < K; k0 += 64) {
        // ---- A tile: 128 rows x 64 k -> f16 LDS (two half16 stores) -------
        {
            half16 h0 = {}, h1 = {};
            if (aval) {
                if (MODE == 2) {
                    const _Float16* p = Af16 + asrc * K + k0 + akseg;
                    h0 = *(const half16*)p;
                    h1 = *(const half16*)(p + 16);
                } else {
                    const float4* p = (const float4*)(Af32 + asrc * K + k0 + akseg);
                    h0 = cvt16(p);
                    h1 = cvt16(p + 4);
                }
            }
            *(half16*)&As[arow_ld * 64 + akseg]      = h0;
            *(half16*)&As[arow_ld * 64 + akseg + 16] = h1;
        }
        // ---- B tile: 64 k x 128 n -> transposed f16 LDS, packed b32 -------
        // thread owns k rows {2q,2q+1, 2q+32,2q+33} x 8 columns
        {
            #pragma unroll
            for (int g = 0; g < 2; ++g) {
                const int k = 2 * bkq + 32 * g;
                const float* r0 = W + (size_t)(k0 + k)     * Ncols + n0 + bcol8;
                const float* r1 = W + (size_t)(k0 + k + 1) * Ncols + n0 + bcol8;
                F4 a0, a1, b0, b1;
                a0.v = ((const float4*)r0)[0];  a1.v = ((const float4*)r0)[1];
                b0.v = ((const float4*)r1)[0];  b1.v = ((const float4*)r1)[1];
                #pragma unroll
                for (int j = 0; j < 4; ++j) {
                    h2pack p0; p0[0] = (_Float16)a0.s[j]; p0[1] = (_Float16)b0.s[j];
                    h2pack p1; p1[0] = (_Float16)a1.s[j]; p1[1] = (_Float16)b1.s[j];
                    *(h2pack*)&Bs[(bcol8 + j)     * 64 + k] = p0;
                    *(h2pack*)&Bs[(bcol8 + j + 4) * 64 + k] = p1;
                }
            }
            if (k0 + 64 < K)   // next weight tile -> L2 (global_prefetch_b8)
                __builtin_prefetch(W + (size_t)(k0 + 64 + 2 * bkq) * Ncols + n0 + bcol8, 0, 1);
        }
        __syncthreads();

        // ---- two k-substeps of 32: 8 WMMAs each -------------------------
        const int koffA = (lane >> 4) * 8;
        const int koffB = (lane >> 4) * 8;
        #pragma unroll
        for (int ks = 0; ks < 64; ks += 32) {
            // A 16-bit 16x32: lane<16 -> row=lane,    K {0..7,16..23}
            //                 lane>=16-> row=lane-16, K {8..15,24..31}
            half16 afr[4];
            #pragma unroll
            for (int mt = 0; mt < 4; ++mt) {
                const int ar = wm * 64 + mt * 16 + (lane & 15);
                AFrag u;
                u.h[0] = *(const half8*)&As[ar * 64 + ks + koffA];
                u.h[1] = *(const half8*)&As[ar * 64 + ks + koffA + 16];
                afr[mt] = u.v;
            }
            // B 16-bit 32x16: lane<16 -> col=lane, K {0..7,16..23} ; lane>=16 -> K {8..15,24..31}
            half16 bfr[2];
            #pragma unroll
            for (int nt = 0; nt < 2; ++nt) {
                const int bc = wn * 32 + nt * 16 + (lane & 15);
                AFrag u;
                u.h[0] = *(const half8*)&Bs[bc * 64 + ks + koffB];
                u.h[1] = *(const half8*)&Bs[bc * 64 + ks + koffB + 16];
                bfr[nt] = u.v;
            }
            #pragma unroll
            for (int mt = 0; mt < 4; ++mt)
                #pragma unroll
                for (int nt = 0; nt < 2; ++nt)
                    acc[mt][nt] = __builtin_amdgcn_wmma_f32_16x16x32_f16(
                        false, afr[mt], false, bfr[nt],
                        (short)0, acc[mt][nt], false, false);
        }
        __syncthreads();
    }

    // ---- epilogue: C/D layout M = r + 8*(lane>=16), N = lane&15 ------------
    const int mhi = (lane >> 4) * 8;
    float bcolv[2];
    int   colv[2];
    #pragma unroll
    for (int nt = 0; nt < 2; ++nt) {
        colv[nt]  = n0 + wn * 32 + nt * 16 + (lane & 15);
        bcolv[nt] = bias[colv[nt]];
    }
    #pragma unroll
    for (int mt = 0; mt < 4; ++mt) {
        #pragma unroll
        for (int r = 0; r < 8; ++r) {
            const int lrow = m0 + wm * 64 + mt * 16 + r + mhi;
            if (lrow >= cnt) continue;
            if (MODE == 2) {
                const int   ent = entries[ebase + lrow];  // token*2+slot
                const float w   = entry_w[ebase + lrow];
                #pragma unroll
                for (int nt = 0; nt < 2; ++nt) {
                    const float v = acc[mt][nt][r] + bcolv[nt];
                    const float s = 1.0f / (1.0f + expf(-v));
                    Out32[(size_t)ent * Ncols + colv[nt]] = s * w;
                }
            } else {
                #pragma unroll
                for (int nt = 0; nt < 2; ++nt) {
                    const float v = acc[mt][nt][r] + bcolv[nt];
                    Out16[(size_t)lrow * Ncols + colv[nt]] = (_Float16)gelu_exact(v);
                }
            }
        }
    }
}

// ---------------------------------------------------------------------------
// Gate layer 2 + softmax-top2 + routing.  One 64-thread block per token.
// Normalized top2 weights: w0 = 1/(1+e^{l1-l0}), w1 = 1-w0 (softmax Z cancels)
// ---------------------------------------------------------------------------
__global__ __launch_bounds__(64)
void moe_gate2_kernel(const _Float16* __restrict__ gh,   // [N, 768] f16
                      const float* __restrict__ w2,      // [768, 12]
                      const float* __restrict__ b2,      // [12]
                      int* __restrict__ counts,
                      int* __restrict__ entries,
                      float* __restrict__ entry_w)
{
    const int n = blockIdx.x;
    const int t = threadIdx.x;
    float acc[NEXP];
    #pragma unroll
    for (int e = 0; e < NEXP; ++e) acc[e] = 0.0f;

    for (int i = t; i < DIM_E; i += 64) {
        const float x = (float)gh[(size_t)n * DIM_E + i];
        const float* wr = w2 + (size_t)i * NEXP;
        #pragma unroll
        for (int e = 0; e < NEXP; ++e) acc[e] += x * wr[e];
    }

    __shared__ float red[64][NEXP];
    #pragma unroll
    for (int e = 0; e < NEXP; ++e) red[t][e] = acc[e];
    __syncthreads();

    if (t == 0) {
        float lg[NEXP];
        #pragma unroll
        for (int e = 0; e < NEXP; ++e) {
            float s = 0.0f;
            for (int j = 0; j < 64; ++j) s += red[j][e];
            lg[e] = s + b2[e];
        }
        int i0 = 0;
        #pragma unroll
        for (int e = 1; e < NEXP; ++e) if (lg[e] > lg[i0]) i0 = e;
        int i1 = (i0 == 0) ? 1 : 0;
        #pragma unroll
        for (int e = 0; e < NEXP; ++e)
            if (e != i0 && lg[e] > lg[i1]) i1 = e;
        const float e1 = expf(lg[i1] - lg[i0]);
        const float w0 = 1.0f / (1.0f + e1);
        const float w1 = e1 / (1.0f + e1);
        const int p0 = atomicAdd(&counts[i0], 1);
        entries[i0 * TOK_N + p0] = n * 2 + 0;
        entry_w[i0 * TOK_N + p0] = w0;
        const int p1 = atomicAdd(&counts[i1], 1);
        entries[i1 * TOK_N + p1] = n * 2 + 1;
        entry_w[i1 * TOK_N + p1] = w1;
    }
}

// ---------------------------------------------------------------------------
// Combine the two weighted slot rows per token: fused[n] = slot[2n] + slot[2n+1]
// ---------------------------------------------------------------------------
__global__ __launch_bounds__(256)
void moe_combine_kernel(const float* __restrict__ slot, float* __restrict__ out)
{
    const int idx = blockIdx.x * 256 + threadIdx.x;
    if (idx >= TOK_N * DIM_E) return;
    const int n = idx / DIM_E;
    const int c = idx - n * DIM_E;
    out[idx] = slot[(size_t)(2 * n) * DIM_E + c] +
               slot[(size_t)(2 * n + 1) * DIM_E + c];
}

// ---------------------------------------------------------------------------
// Host-side orchestration (graph-capture safe: only async launches/memset)
// ---------------------------------------------------------------------------
extern "C" void kernel_launch(void* const* d_in, const int* in_sizes, int n_in,
                              void* d_out, int out_size, void* d_ws, size_t ws_size,
                              hipStream_t stream)
{
    const float* combined = (const float*)d_in[0];  // [8192, 1536]
    const float* gate_w1  = (const float*)d_in[1];  // [1536, 768]
    const float* gate_b1  = (const float*)d_in[2];  // [768]
    const float* gate_w2  = (const float*)d_in[3];  // [768, 12]
    const float* gate_b2  = (const float*)d_in[4];  // [12]
    const float* ew1      = (const float*)d_in[5];  // [12, 1536, 3072]
    const float* eb1      = (const float*)d_in[6];  // [12, 3072]
    const float* ew2      = (const float*)d_in[7];  // [12, 3072, 768]
    const float* eb2      = (const float*)d_in[8];  // [12, 768]
    float* out = (float*)d_out;                     // [8192, 768]

    // workspace layout (bytes)
    char* ws = (char*)d_ws;
    size_t off = 0;
    _Float16* gh = (_Float16*)(ws + off);  off += (size_t)TOK_N * DIM_E * 2;      // 12.6 MB
    off = (off + 255) & ~(size_t)255;
    int* counts = (int*)(ws + off);        off += 256;
    int* entries = (int*)(ws + off);       off += (size_t)NEXP * TOK_N * 4;       // 0.39 MB
    float* entry_w = (float*)(ws + off);   off += (size_t)NEXP * TOK_N * 4;       // 0.39 MB
    off = (off + 255) & ~(size_t)255;
    _Float16* H = (_Float16*)(ws + off);   off += (size_t)TOK_N * DIM_4E * 2;     // 50.3 MB
    off = (off + 255) & ~(size_t)255;
    float* slot = (float*)(ws + off);      off += (size_t)TOK_N * 2 * DIM_E * 4;  // 50.3 MB

    hipMemsetAsync(counts, 0, 256, stream);

    // Gate layer 1: gh = gelu(combined @ gate_w1 + b1)    M=8192 K=1536 N=768
    moe_gemm_kernel<0><<<dim3(DIM_E / 128, TOK_N / 128), 256, 0, stream>>>(
        combined, nullptr, gate_w1, gate_b1, gh, nullptr,
        nullptr, nullptr, nullptr, 0, TOK_N, DIM_2E, DIM_E);

    // Gate layer 2 + top-2 routing
    moe_gate2_kernel<<<TOK_N, 64, 0, stream>>>(gh, gate_w2, gate_b2,
                                               counts, entries, entry_w);

    // Per-expert sparse MLP (stream-serialized; H reused across experts)
    for (int e = 0; e < NEXP; ++e) {
        moe_gemm_kernel<1><<<dim3(DIM_4E / 128, TOK_N / 128), 256, 0, stream>>>(
            combined, nullptr,
            ew1 + (size_t)e * DIM_2E * DIM_4E, eb1 + (size_t)e * DIM_4E,
            H, nullptr, entries, entry_w, counts, e, 0, DIM_2E, DIM_4E);
        moe_gemm_kernel<2><<<dim3(DIM_E / 128, TOK_N / 128), 256, 0, stream>>>(
            nullptr, H,
            ew2 + (size_t)e * DIM_4E * DIM_E, eb2 + (size_t)e * DIM_E,
            nullptr, slot, entries, entry_w, counts, e, 0, DIM_4E, DIM_E);
    }

    // fused[n] = slot[2n] + slot[2n+1]
    moe_combine_kernel<<<(TOK_N * DIM_E) / 256, 256, 0, stream>>>(slot, out);
}
